// CrossAttention_5342939316448
// MI455X (gfx1250) — compile-verified, matrix operand order checked by reading, not run on hardware
//
#include <hip/hip_runtime.h>
#include <hip/hip_bf16.h>

typedef __attribute__((ext_vector_type(16))) _Float16 v16h;
typedef __attribute__((ext_vector_type(8)))  float    v8f;

#define B_   16
#define SQ   4096
#define DM   512
#define SKV  77
#define DC   768
#define NH   8
#define DH   64
#define INNER 512
#define JPAD 128   // Skv padded to multiple of 32 (K-dim of attn@v WMMA)

// ---------------------------------------------------------------------------
// helpers
// ---------------------------------------------------------------------------
__device__ __forceinline__ v8f wmma32(v16h a, v16h b, v8f c) {
    // D = A(16x32 f16) * B(32x16 f16) + C(16x16 f32)
    return __builtin_amdgcn_wmma_f32_16x16x32_f16(
        /*neg_a=*/false, a, /*neg_b=*/false, b,
        /*c_mod=*/(short)0, c, /*reuse_a=*/false, /*reuse_b=*/false);
}

// One fragment load works for both A and B operands:
//   A: base = &mat[row0*stride + k0]     lane picks m = lane&15, 16 halfs along K
//   B: base = &W[n0*K + k0] (N-major)    lane picks n = lane&15, 16 halfs along K
// Per-lane 32-byte contiguous load -> b128 pairs (ds_load / global_load).
__device__ __forceinline__ v16h frag16(const _Float16* base, int stride, int lane) {
    return *(const v16h*)(base + (size_t)(lane & 15) * stride + ((lane >> 4) << 4));
}

// ---------------------------------------------------------------------------
// Kernel 0: convert f32 weight [K][N] -> f16 transposed [N][K]
// ---------------------------------------------------------------------------
__global__ void wcvt_t(const float* __restrict__ src, _Float16* __restrict__ dst,
                       int K, int N) {
    int idx = blockIdx.x * 256 + threadIdx.x;
    if (idx < K * N) {
        int k = idx / N, n = idx - k * N;
        dst[(size_t)n * K + k] = (_Float16)src[idx];
    }
}

// ---------------------------------------------------------------------------
// Kernel 1: k/v projection (WMMA).
//   context[b][j][768] @ Wk/Wv -> per-head layouts:
//     kP[b][h][j(128)][d(64)]   (B-operand of scores:  K=d, N=j -> [j][d] N-major)
//     vT[b][h][d(64)][j(128)]   (B-operand of attn@v:  K=j, N=d -> [d][j] N-major)
// grid = (5 m-tiles, 16 batches, 2 {k,v}), block = 256 (8 waves)
// ---------------------------------------------------------------------------
__global__ __launch_bounds__(256) void kv_proj(
    const float* __restrict__ ctx,
    const _Float16* __restrict__ WkT,   // [512][768] f16
    const _Float16* __restrict__ WvT,   // [512][768] f16
    _Float16* __restrict__ kP,
    _Float16* __restrict__ vT) {
    __shared__ __align__(32) _Float16 ct[16 * DC];   // 24 KB context tile (f16)

    const int tid  = threadIdx.x;
    const int b    = blockIdx.y;
    const int j0   = blockIdx.x * 16;
    const int isv  = blockIdx.z;
    const int w    = tid >> 5, lane = tid & 31;

    const float* cbase = ctx + (size_t)b * SKV * DC;
    for (int i = tid; i < 16 * DC; i += 256) {
        int r = i / DC, c = i - r * DC;
        int j = j0 + r;
        ct[i] = (j < SKV) ? (_Float16)cbase[(size_t)j * DC + c] : (_Float16)0.f;
    }
    __syncthreads();

    const _Float16* WT = isv ? WvT : WkT;
    const int n0 = w * 64;

    v8f acc[4] = {v8f{}, v8f{}, v8f{}, v8f{}};
    for (int ks = 0; ks < DC / 32; ++ks) {
        const int k0 = ks * 32;
        v16h a = frag16(ct + k0, DC, lane);
#pragma unroll
        for (int t = 0; t < 4; ++t) {
            v16h bf = frag16(WT + (size_t)(n0 + t * 16) * DC + k0, DC, lane);
            acc[t] = wmma32(a, bf, acc[t]);
        }
    }

#pragma unroll
    for (int t = 0; t < 4; ++t) {
        const int n = n0 + t * 16 + (lane & 15);
        const int h = n >> 6, d = n & 63;
#pragma unroll
        for (int g = 0; g < 8; ++g) {
            const int j = j0 + g + 8 * (lane >> 4);
            if (j < SKV) {
                _Float16 val = (_Float16)acc[t][g];
                if (isv) vT[(((size_t)b * NH + h) * DH + d) * JPAD + j] = val;
                else     kP[(((size_t)b * NH + h) * JPAD + j) * DH + d] = val;
            }
        }
    }
}

// ---------------------------------------------------------------------------
// Kernel 2: fused q-proj + attention + out-proj.
// One block = 64 q rows of one batch. 256 threads = 8 waves.
// bufA: x tile (f16) then h tile; bufB: q tile (f16, pre-scaled by d^-1/2).
// ---------------------------------------------------------------------------
__global__ __launch_bounds__(256) void attn_fused(
    const float*    __restrict__ x,
    const _Float16* __restrict__ WqT,    // [512][512]
    const _Float16* __restrict__ kP,     // [B][H][128][64]
    const _Float16* __restrict__ vT,     // [B][H][64][128]
    const _Float16* __restrict__ WoT,    // [512][512]
    const float*    __restrict__ bout,   // [512]
    float*          __restrict__ out) {
    __shared__ __align__(32) _Float16 bufA[64 * INNER];  // 64 KB  x -> h
    __shared__ __align__(32) _Float16 bufB[64 * INNER];  // 64 KB  q
    __shared__ __align__(32) float    sc[64 * JPAD];     // 32 KB  scores
    __shared__ __align__(32) _Float16 wb[64 * JPAD];     // 16 KB  softmax weights

    const int tid  = threadIdx.x;
    const int w    = tid >> 5, lane = tid & 31;
    const int b    = blockIdx.y;
    const int m0   = blockIdx.x * 64;

    // ---------- phase 0: x tile -> f16 LDS (tile is one contiguous block) ----
    const float4* xs = (const float4*)(x + ((size_t)b * SQ + m0) * DM);
    __builtin_prefetch(WqT + (size_t)tid * 1024, 0, 1);   // warm L2 with Wq
    for (int i = tid; i < 64 * DM / 4; i += 256) {
        float4 v = xs[i];
        _Float16* d = bufA + (size_t)i * 4;
        d[0] = (_Float16)v.x; d[1] = (_Float16)v.y;
        d[2] = (_Float16)v.z; d[3] = (_Float16)v.w;
    }
    __syncthreads();

    // ---------- phase 1: q = x @ Wq  (scaled), write f16 to bufB -------------
    {
        const int wm = w & 3, wn = w >> 2;                 // 4 x 2 wave grid
        const _Float16* arow = bufA + (size_t)(wm * 16) * INNER;
        for (int nc = 0; nc < 4; ++nc) {
            const int n0 = wn * 256 + nc * 64;
            v8f q[4] = {v8f{}, v8f{}, v8f{}, v8f{}};
            for (int ks = 0; ks < INNER / 32; ++ks) {
                const int k0 = ks * 32;
                v16h a = frag16(arow + k0, INNER, lane);
#pragma unroll
                for (int t = 0; t < 4; ++t) {
                    v16h bf = frag16(WqT + (size_t)(n0 + t * 16) * INNER + k0,
                                     INNER, lane);
                    q[t] = wmma32(a, bf, q[t]);
                }
            }
#pragma unroll
            for (int t = 0; t < 4; ++t) {
                const int n = n0 + t * 16 + (lane & 15);
#pragma unroll
                for (int g = 0; g < 8; ++g) {
                    const int m = wm * 16 + g + 8 * (lane >> 4);
                    bufB[(size_t)m * INNER + n] = (_Float16)(q[t][g] * 0.125f);
                }
            }
        }
    }
    __syncthreads();

    // ---------- phase 2: per-head scores -> softmax -> attn@v ----------------
    for (int h = 0; h < NH; ++h) {
        {   // scores[64][128] = q_h (64x64) @ k_h^T, waves: 4(m) x 2(j-half)
            const int sm = w & 3, sj = w >> 2;
            const _Float16* qrow = bufB + (size_t)(sm * 16) * INNER + h * DH;
            const _Float16* kb   = kP + (((size_t)b * NH + h) * JPAD) * DH;
            v8f s[4] = {v8f{}, v8f{}, v8f{}, v8f{}};
#pragma unroll
            for (int dstep = 0; dstep < 2; ++dstep) {
                const int d0 = dstep * 32;
                v16h a = frag16(qrow + d0, INNER, lane);
#pragma unroll
                for (int t = 0; t < 4; ++t) {
                    const int j0 = sj * 64 + t * 16;
                    v16h bf = frag16(kb + (size_t)j0 * DH + d0, DH, lane);
                    s[t] = wmma32(a, bf, s[t]);
                }
            }
#pragma unroll
            for (int t = 0; t < 4; ++t) {
                const int j = sj * 64 + t * 16 + (lane & 15);
#pragma unroll
                for (int g = 0; g < 8; ++g) {
                    const int m = sm * 16 + g + 8 * (lane >> 4);
                    sc[m * JPAD + j] = s[t][g];
                }
            }
        }
        __syncthreads();

        if (tid < 64) {  // one thread per q row: softmax over 77 keys
            float* row = sc + tid * JPAD;
            float mx = -1e30f;
            for (int j = 0; j < SKV; ++j) mx = fmaxf(mx, row[j]);
            float sum = 0.f;
            for (int j = 0; j < SKV; ++j) {
                float e = __expf(row[j] - mx);
                row[j] = e; sum += e;
            }
            const float r = 1.f / sum;
            _Float16* wr = wb + tid * JPAD;
            for (int j = 0; j < SKV; ++j)    wr[j] = (_Float16)(row[j] * r);
            for (int j = SKV; j < JPAD; ++j) wr[j] = (_Float16)0.f;
        }
        __syncthreads();

        {   // attn (64x64) = w (64x128) @ v_h (128x64), waves: 4(m) x 2(d-half)
            const int am = w & 3, ad = w >> 2;
            const _Float16* wrow = wb + (size_t)(am * 16) * JPAD;
            const _Float16* vb   = vT + (((size_t)b * NH + h) * DH) * JPAD;
            v8f av[2] = {v8f{}, v8f{}};
#pragma unroll
            for (int js = 0; js < 4; ++js) {
                const int j0 = js * 32;
                v16h a = frag16(wrow + j0, JPAD, lane);
#pragma unroll
                for (int t = 0; t < 2; ++t) {
                    const int d0 = ad * 32 + t * 16;
                    v16h bf = frag16(vb + (size_t)d0 * JPAD + j0, JPAD, lane);
                    av[t] = wmma32(a, bf, av[t]);
                }
            }
#pragma unroll
            for (int t = 0; t < 2; ++t) {
                const int d = ad * 32 + t * 16 + (lane & 15);
#pragma unroll
                for (int g = 0; g < 8; ++g) {
                    const int m = am * 16 + g + 8 * (lane >> 4);
                    bufA[(size_t)m * INNER + h * DH + d] = (_Float16)av[t][g];
                }
            }
        }
        __syncthreads();
    }

    // ---------- phase 3: out = h @ Wout + bout -------------------------------
    {
        const int wm = w & 3, wn = w >> 2;
        const _Float16* hrow = bufA + (size_t)(wm * 16) * INNER;
        float* orow = out + ((size_t)b * SQ + m0 + wm * 16) * INNER;
        for (int nc = 0; nc < 4; ++nc) {
            const int n0 = wn * 256 + nc * 64;
            v8f o[4] = {v8f{}, v8f{}, v8f{}, v8f{}};
            for (int ks = 0; ks < INNER / 32; ++ks) {
                const int k0 = ks * 32;
                v16h a = frag16(hrow + k0, INNER, lane);
#pragma unroll
                for (int t = 0; t < 4; ++t) {
                    v16h bf = frag16(WoT + (size_t)(n0 + t * 16) * INNER + k0,
                                     INNER, lane);
                    o[t] = wmma32(a, bf, o[t]);
                }
            }
#pragma unroll
            for (int t = 0; t < 4; ++t) {
                const int n = n0 + t * 16 + (lane & 15);
                const float bias = bout[n];
#pragma unroll
                for (int g = 0; g < 8; ++g) {
                    orow[(size_t)(g + 8 * (lane >> 4)) * INNER + n] = o[t][g] + bias;
                }
            }
        }
    }
}

// ---------------------------------------------------------------------------
// launch
// ---------------------------------------------------------------------------
extern "C" void kernel_launch(void* const* d_in, const int* in_sizes, int n_in,
                              void* d_out, int out_size, void* d_ws, size_t ws_size,
                              hipStream_t stream) {
    const float* x    = (const float*)d_in[0];
    const float* ctx  = (const float*)d_in[1];
    const float* Wq   = (const float*)d_in[2];
    const float* Wk   = (const float*)d_in[3];
    const float* Wv   = (const float*)d_in[4];
    const float* Wout = (const float*)d_in[5];
    const float* bout = (const float*)d_in[6];
    float* out = (float*)d_out;

    char* ws = (char*)d_ws;
    size_t off = 0;
    _Float16* WqT = (_Float16*)(ws + off); off += (size_t)INNER * DM * 2;     // [512][512]
    _Float16* WkT = (_Float16*)(ws + off); off += (size_t)INNER * DC * 2;     // [512][768]
    _Float16* WvT = (_Float16*)(ws + off); off += (size_t)INNER * DC * 2;     // [512][768]
    _Float16* WoT = (_Float16*)(ws + off); off += (size_t)INNER * INNER * 2;  // [512][512]
    _Float16* kP  = (_Float16*)(ws + off);
    size_t kvBytes = 2ull * B_ * NH * JPAD * DH * 2;                          // kP + vT
    _Float16* vT  = (_Float16*)(ws + off + kvBytes / 2);

    // zero the j-padding of kP/vT (graph-capturable)
    hipMemsetAsync(kP, 0, kvBytes, stream);

    // weight convert + transpose to f16 [N][K]
    wcvt_t<<<dim3((DM * INNER + 255) / 256), 256, 0, stream>>>(Wq, WqT, DM, INNER);
    wcvt_t<<<dim3((DC * INNER + 255) / 256), 256, 0, stream>>>(Wk, WkT, DC, INNER);
    wcvt_t<<<dim3((DC * INNER + 255) / 256), 256, 0, stream>>>(Wv, WvT, DC, INNER);
    wcvt_t<<<dim3((INNER * INNER + 255) / 256), 256, 0, stream>>>(Wout, WoT, INNER, INNER);

    // k/v projection: 5 j-tiles x 16 batches x {k,v}
    kv_proj<<<dim3(5, B_, 2), 256, 0, stream>>>(ctx, WkT, WvT, kP, vT);

    // fused q-proj + attention + out-proj: 64 q-tiles x 16 batches
    attn_fused<<<dim3(SQ / 64, B_), 256, 0, stream>>>(x, WqT, kP, vT, WoT, bout, out);
}
